// ProfilingMoELayer_64183991271941
// MI455X (gfx1250) — matrix operand order repaired and checked by reading, not compiled, not run on hardware
//
#include <hip/hip_runtime.h>
#include <hip/hip_bf16.h>
#include <math.h>

// ---------------- problem constants (from reference) ----------------
#define NE   8          // experts
#define TOPK 2
#define HD   1024       // hidden
#define FD   4096       // intermediate
#define NT   8192       // tokens (B*S = 2*4096)
#define TK   (NT*TOPK)  // 16384 assignments
#define CAP  2560       // int(T*K/E * 1.25)

typedef __attribute__((ext_vector_type(16))) __bf16 v16bf;
typedef __attribute__((ext_vector_type(8)))  float  v8f;
typedef __attribute__((ext_vector_type(4)))  float  f32x4;
typedef __attribute__((ext_vector_type(4)))  __bf16 v4bf;
typedef __attribute__((ext_vector_type(2)))  __bf16 v2bf;

union FragAB { v16bf v; f32x4 f4[2]; };

// =====================================================================
// 1) Gate: logits = x @ gate_w + gate_b ; top-2 ; pair softmax ;
//    accumulate full softmax sums for aux loss.
//    One wave (32 lanes) per token; 8 waves / block.
// =====================================================================
__global__ __launch_bounds__(256) void moe_gate(const float* __restrict__ x,
                                                const float* __restrict__ gw,
                                                const float* __restrict__ gb,
                                                int*   __restrict__ a_exp,
                                                float* __restrict__ a_prob,
                                                float* __restrict__ gate_sum)
{
    __shared__ float ssum[NE];
    const int tid  = threadIdx.x;
    if (tid < NE) ssum[tid] = 0.f;
    __syncthreads();

    const int wv   = tid >> 5;
    const int lane = tid & 31;
    const int t    = blockIdx.x * 8 + wv;
    const float* xr = x + (size_t)t * HD;

    float acc[NE] = {0.f,0.f,0.f,0.f,0.f,0.f,0.f,0.f};
    for (int c = 0; c < HD / 32; ++c) {
        const int h = c * 32 + lane;
        const float xv = xr[h];
        const float* g = gw + (size_t)h * NE;
        f32x4 g0 = *(const f32x4*)g;
        f32x4 g1 = *(const f32x4*)(g + 4);
        acc[0] += xv * g0.x;  acc[1] += xv * g0.y;
        acc[2] += xv * g0.z;  acc[3] += xv * g0.w;
        acc[4] += xv * g1.x;  acc[5] += xv * g1.y;
        acc[6] += xv * g1.z;  acc[7] += xv * g1.w;
    }
    #pragma unroll
    for (int e = 0; e < NE; ++e) {
        #pragma unroll
        for (int m = 16; m >= 1; m >>= 1) acc[e] += __shfl_xor(acc[e], m, 32);
    }

    if (lane == 0) {
        #pragma unroll
        for (int e = 0; e < NE; ++e) acc[e] += gb[e];
        // top-2 (strict > keeps lowest index on ties, like jax top_k)
        int i0 = 0; float v0 = acc[0];
        #pragma unroll
        for (int e = 1; e < NE; ++e) if (acc[e] > v0) { v0 = acc[e]; i0 = e; }
        int i1 = -1; float v1 = -3.4e38f;
        #pragma unroll
        for (int e = 0; e < NE; ++e) if (e != i0 && acc[e] > v1) { v1 = acc[e]; i1 = e; }
        a_exp[2*t]   = i0;  a_exp[2*t+1] = i1;
        // softmax over the two top logits (TEMP = 1)
        const float e1  = __expf(v1 - v0);
        const float inv = 1.f / (1.f + e1);
        a_prob[2*t]   = inv;
        a_prob[2*t+1] = e1 * inv;
        // full softmax for the aux loss mean
        float mx = acc[0];
        #pragma unroll
        for (int e = 1; e < NE; ++e) mx = fmaxf(mx, acc[e]);
        float ex[NE], s = 0.f;
        #pragma unroll
        for (int e = 0; e < NE; ++e) { ex[e] = __expf(acc[e] - mx); s += ex[e]; }
        const float is = 1.f / s;
        #pragma unroll
        for (int e = 0; e < NE; ++e) atomicAdd(&ssum[e], ex[e] * is);
    }
    __syncthreads();
    if (tid < NE) atomicAdd(&gate_sum[tid], ssum[tid]);
}

// =====================================================================
// 2) Sequential capacity assignment via parallel prefix sum.
//    One workgroup, 1024 threads, 16 assignments each.
// =====================================================================
__global__ __launch_bounds__(1024) void moe_scan(const int* __restrict__ a_exp,
                                                 int* __restrict__ dest,
                                                 int* __restrict__ slot_src,
                                                 int* __restrict__ counts)
{
    __shared__ int s[1024 * 9];     // padded to 9 ints/thread to spread banks
    const int t = threadIdx.x;

    for (int i = t; i < NE * CAP; i += 1024) slot_src[i] = -1;

    int myexp[16];
    int own[NE] = {0,0,0,0,0,0,0,0};
    #pragma unroll
    for (int j = 0; j < 16; ++j) {
        myexp[j] = a_exp[t * 16 + j];
        #pragma unroll
        for (int e = 0; e < NE; ++e) own[e] += (myexp[j] == e) ? 1 : 0;
    }
    #pragma unroll
    for (int e = 0; e < NE; ++e) s[t * 9 + e] = own[e];
    __syncthreads();

    for (int off = 1; off < 1024; off <<= 1) {
        int v[NE];
        #pragma unroll
        for (int e = 0; e < NE; ++e) v[e] = (t >= off) ? s[(t - off) * 9 + e] : 0;
        __syncthreads();
        #pragma unroll
        for (int e = 0; e < NE; ++e) s[t * 9 + e] += v[e];
        __syncthreads();
    }

    int base[NE];
    #pragma unroll
    for (int e = 0; e < NE; ++e) base[e] = s[t * 9 + e] - own[e];   // exclusive
    if (t == 1023) {
        #pragma unroll
        for (int e = 0; e < NE; ++e) counts[e] = s[t * 9 + e];       // totals
    }
    #pragma unroll
    for (int j = 0; j < 16; ++j) {
        const int i  = t * 16 + j;
        const int ej = myexp[j];
        int pos = 0;
        #pragma unroll
        for (int e = 0; e < NE; ++e) if (ej == e) pos = base[e]++;
        if (pos < CAP) {
            dest[i] = ej * CAP + pos;
            slot_src[ej * CAP + pos] = i >> 1;   // source token
        } else {
            dest[i] = -1;                        // dropped (overflow)
        }
    }
}

// =====================================================================
// 3) Pack expert_in (bf16, row-major [slot][H]); zero unused slots.
// =====================================================================
__global__ __launch_bounds__(256) void moe_pack(const float* __restrict__ x,
                                                const int* __restrict__ slot_src,
                                                __bf16* __restrict__ ein)
{
    const int slot = blockIdx.x;
    const int tok  = slot_src[slot];
    const int h4   = threadIdx.x * 4;
    v4bf o;
    if (tok >= 0) {
        f32x4 v = *(const f32x4*)(x + (size_t)tok * HD + h4);
        o.x = (__bf16)v.x; o.y = (__bf16)v.y; o.z = (__bf16)v.z; o.w = (__bf16)v.w;
    } else {
        o.x = (__bf16)0.f; o.y = (__bf16)0.f; o.z = (__bf16)0.f; o.w = (__bf16)0.f;
    }
    *(v4bf*)(ein + (size_t)slot * HD + h4) = o;
}

// =====================================================================
// 4) Expert GEMM: C[CAP x Nd] = A[CAP x Kd](bf16) @ W[Kd x Nd](fp32->bf16)
//    per expert, + bias, optional exact GELU.
//    Block tile 128M x 256N x 32K, 8 waves in a 2x4 grid, wave tile
//    64x64 = 4x4 WMMA tiles (16 v_wmma per K-panel, 16 ds_load_b128).
//    W is converted + transposed into LDS as [N][K] bf16; two adjacent
//    K-rows are packed per 32-bit LDS store (v_cvt_pk_bf16_f32 pairs).
//    Fragment layouts follow CDNA5 ISA 7.12.2 exactly.
// =====================================================================
#define LDS_ROW      80                   // 32 bf16 (64B) + 16B pad
#define A_TILE_BYTES (128 * LDS_ROW)      // 10240
#define B_TILE_BYTES (256 * LDS_ROW)      // 20480
#define BUF_BYTES    (A_TILE_BYTES + B_TILE_BYTES)

template<int Kd, int Nd, bool GELU>
__global__ __launch_bounds__(256) void moe_gemm(const __bf16* __restrict__ A,
                                                const float*  __restrict__ Bw,
                                                const float*  __restrict__ bias,
                                                __bf16* __restrict__ outB,
                                                float*  __restrict__ outF)
{
    __shared__ __attribute__((aligned(16))) unsigned char lds[2 * BUF_BYTES];

    const int tid   = threadIdx.x;
    const int e     = blockIdx.z;
    const int mBase = blockIdx.y * 128;
    const int nBase = blockIdx.x * 256;

    const __bf16* Ae    = A    + (size_t)e * CAP * Kd;
    const float*  Be    = Bw   + (size_t)e * (size_t)Kd * Nd;
    const float*  biasE = bias + (size_t)e * Nd;

    const int lane = tid & 31;
    const int w8   = tid >> 5;
    const int wr   = w8 >> 2;          // 0..1  (M, 64 each)
    const int wc   = w8 & 3;           // 0..3  (N, 64 each)
    const int half = lane >> 4;
    const int l16  = lane & 15;

    auto loadA = [&](int buf, int kBase) {
        unsigned char* dst = lds + buf * BUF_BYTES;
        #pragma unroll
        for (int i = 0; i < 2; ++i) {
            const int c   = tid + i * 256;        // 512 16B chunks
            const int row = c >> 2;
            const int off = (c & 3) * 16;
            const unsigned char* src =
                (const unsigned char*)(Ae + (size_t)(mBase + row) * Kd + kBase) + off;
            if (i == 0 && kBase + 64 < Kd)
                __builtin_prefetch(src + 128, 0, 1);               // 2 K-tiles ahead
            f32x4 v = *(const f32x4*)src;
            *(f32x4*)(dst + row * LDS_ROW + off) = v;
        }
    };
    auto loadB = [&](int buf, int kBase) {
        unsigned char* dst = lds + buf * BUF_BYTES + A_TILE_BYTES;
        #pragma unroll
        for (int i = 0; i < 4; ++i) {
            const int idx = tid + i * 256;        // 1024 (k-pair, n4) units
            const int k0  = (idx >> 6) * 2;       // 0,2,..,30
            const int n4  = (idx & 63) * 4;       // 0..252
            const float* s0 = Be + (size_t)(kBase + k0) * Nd + nBase + n4;
            f32x4 r0 = *(const f32x4*)s0;
            f32x4 r1 = *(const f32x4*)(s0 + Nd);
            if (i == 0 && kBase + 64 < Kd)
                __builtin_prefetch(s0 + (size_t)64 * Nd, 0, 1);    // global_prefetch_b8
            unsigned char* p = dst + k0 * 2;      // [n][k] transpose, k-pair packed
            v2bf t0; t0.x = (__bf16)r0.x; t0.y = (__bf16)r1.x;
            v2bf t1; t1.x = (__bf16)r0.y; t1.y = (__bf16)r1.y;
            v2bf t2; t2.x = (__bf16)r0.z; t2.y = (__bf16)r1.z;
            v2bf t3; t3.x = (__bf16)r0.w; t3.y = (__bf16)r1.w;
            *(v2bf*)(p + (size_t)(n4 + 0) * LDS_ROW) = t0;
            *(v2bf*)(p + (size_t)(n4 + 1) * LDS_ROW) = t1;
            *(v2bf*)(p + (size_t)(n4 + 2) * LDS_ROW) = t2;
            *(v2bf*)(p + (size_t)(n4 + 3) * LDS_ROW) = t3;
        }
    };

    v8f acc[4][4] = {};

    loadA(0, 0);
    loadB(0, 0);
    __syncthreads();

    int buf = 0;
    const int Ktiles = Kd / 32;
    for (int kk = 0; kk < Ktiles; ++kk) {
        if (kk + 1 < Ktiles) { loadA(buf ^ 1, (kk + 1) * 32); loadB(buf ^ 1, (kk + 1) * 32); }

        const unsigned char* As = lds + buf * BUF_BYTES;
        const unsigned char* Bs = As + A_TILE_BYTES;

        // A fragments: lane<16 -> row=l16, K{0-7,16-23}; lane>=16 -> K{8-15,24-31}
        FragAB a[4], b[4];
        #pragma unroll
        for (int i = 0; i < 4; ++i) {
            const unsigned char* p = As + (wr * 64 + i * 16 + l16) * LDS_ROW + half * 16;
            a[i].f4[0] = *(const f32x4*)p;
            a[i].f4[1] = *(const f32x4*)(p + 32);
        }
        // B fragments: [n][k] layout -> lane<16 K{0-7,16-23}; lane>=16 K{8-15,24-31}
        #pragma unroll
        for (int j = 0; j < 4; ++j) {
            const unsigned char* p = Bs + (wc * 64 + j * 16 + l16) * LDS_ROW + half * 16;
            b[j].f4[0] = *(const f32x4*)p;
            b[j].f4[1] = *(const f32x4*)(p + 32);
        }
        #pragma unroll
        for (int i = 0; i < 4; ++i)
            #pragma unroll
            for (int j = 0; j < 4; ++j)
                acc[i][j] = __builtin_amdgcn_wmma_f32_16x16x32_bf16(
                    false, a[i].v, false, b[j].v, (short)0, acc[i][j], false, false);

        __syncthreads();
        buf ^= 1;
    }

    // Epilogue: C/D layout -> VGPR r, lane half h: (M = r + 8h, N = l16)
    #pragma unroll
    for (int j = 0; j < 4; ++j) {
        const int gn   = nBase + wc * 64 + j * 16 + l16;
        const float bv = biasE[gn];
        #pragma unroll
        for (int i = 0; i < 4; ++i) {
            const int gm0 = mBase + wr * 64 + i * 16 + half * 8;
            #pragma unroll
            for (int r = 0; r < 8; ++r) {
                float v = acc[i][j][r] + bv;
                if (GELU) {
                    const float g = 0.5f * v * (1.0f + erff(v * 0.70710678118654752f));
                    outB[(size_t)e * CAP * Nd + (size_t)(gm0 + r) * Nd + gn] = (__bf16)g;
                } else {
                    outF[(size_t)e * CAP * Nd + (size_t)(gm0 + r) * Nd + gn] = v;
                }
            }
        }
    }
}

// =====================================================================
// 5) Combine: out[t] = sum_k prob_k * eout[dest_k]  (drops contribute 0)
// =====================================================================
__global__ __launch_bounds__(256) void moe_combine(const float* __restrict__ eout,
                                                   const int*   __restrict__ dest,
                                                   const float* __restrict__ a_prob,
                                                   float* __restrict__ out)
{
    const int t  = blockIdx.x;
    const int h4 = threadIdx.x * 4;
    const int d0 = dest[2 * t], d1 = dest[2 * t + 1];
    const float p0 = a_prob[2 * t], p1 = a_prob[2 * t + 1];
    float rx = 0.f, ry = 0.f, rz = 0.f, rw = 0.f;
    if (d0 >= 0) {
        f32x4 v = *(const f32x4*)(eout + (size_t)d0 * HD + h4);
        rx += p0 * v.x; ry += p0 * v.y; rz += p0 * v.z; rw += p0 * v.w;
    }
    if (d1 >= 0) {
        f32x4 v = *(const f32x4*)(eout + (size_t)d1 * HD + h4);
        rx += p1 * v.x; ry += p1 * v.y; rz += p1 * v.z; rw += p1 * v.w;
    }
    f32x4 r; r.x = rx; r.y = ry; r.z = rz; r.w = rw;
    *(f32x4*)(out + (size_t)t * HD + h4) = r;
}

__global__ void moe_aux(const int* __restrict__ counts,
                        const float* __restrict__ gate_sum,
                        float* __restrict__ out_aux)
{
    if (threadIdx.x == 0 && blockIdx.x == 0) {
        float a = 0.f;
        #pragma unroll
        for (int e = 0; e < NE; ++e)
            a += ((float)counts[e] / ((float)TK + 1e-9f)) * (gate_sum[e] / (float)NT);
        a *= (float)NE;
        out_aux[0] = fminf(a * 0.01f, 1.0f);
    }
}

__global__ void moe_init(float* __restrict__ gate_sum)
{
    if (threadIdx.x < NE) gate_sum[threadIdx.x] = 0.f;
}

// =====================================================================
extern "C" void kernel_launch(void* const* d_in, const int* in_sizes, int n_in,
                              void* d_out, int out_size, void* d_ws, size_t ws_size,
                              hipStream_t stream)
{
    (void)in_sizes; (void)n_in; (void)out_size; (void)ws_size;
    const float* x      = (const float*)d_in[0];
    const float* gate_w = (const float*)d_in[1];
    const float* gate_b = (const float*)d_in[2];
    const float* w1     = (const float*)d_in[3];
    const float* b1     = (const float*)d_in[4];
    const float* w2     = (const float*)d_in[5];
    const float* b2     = (const float*)d_in[6];
    float* out = (float*)d_out;

    char* ws = (char*)d_ws;
    int*    a_exp    = (int*)  (ws + 0);
    float*  a_prob   = (float*)(ws + 65536);
    int*    dest     = (int*)  (ws + 131072);
    int*    slot_src = (int*)  (ws + 196608);
    int*    counts   = (int*)  (ws + 278528);
    float*  gate_sum = (float*)(ws + 278784);
    __bf16* ein      = (__bf16*)(ws + (1u << 20));
    __bf16* h1       = (__bf16*)(ws + (1u << 20) + (size_t)NE * CAP * HD * 2);
    float*  eout     = (float*) (ws + (1u << 20) + (size_t)NE * CAP * HD * 2
                                                 + (size_t)NE * CAP * FD * 2);

    moe_init<<<1, 32, 0, stream>>>(gate_sum);
    moe_gate<<<NT / 8, 256, 0, stream>>>(x, gate_w, gate_b, a_exp, a_prob, gate_sum);
    moe_scan<<<1, 1024, 0, stream>>>(a_exp, dest, slot_src, counts);
    moe_pack<<<NE * CAP, 256, 0, stream>>>(x, slot_src, ein);
    moe_gemm<HD, FD, true ><<<dim3(FD / 256, CAP / 128, NE), 256, 0, stream>>>(ein, w1, b1, h1, nullptr);
    moe_gemm<FD, HD, false><<<dim3(HD / 256, CAP / 128, NE), 256, 0, stream>>>(h1, w2, b2, nullptr, eout);
    moe_combine<<<NT, 256, 0, stream>>>(eout, dest, a_prob, out);
    moe_aux<<<1, 32, 0, stream>>>(counts, gate_sum, out + (size_t)NT * HD);
}
